// HybridBlock_20066087207295
// MI455X (gfx1250) — matrix operand order repaired and checked by reading, not compiled, not run on hardware
//
#include <hip/hip_runtime.h>

// ---------------------------------------------------------------------------
// HybridBlock for MI455X (gfx1250, wave32, WMMA, TDM)
//   LayerNorm -> {QKV proj, edge RBF-MLP, edge attention} -> scatter-add -> out proj
// GEMMs use v_wmma_f32_16x16x32_bf16; projection kernels stage their A-strip
// into LDS with tensor_load_to_lds (TDM, hardware pad insertion), synced via
// s_wait_tensorcnt.
// ---------------------------------------------------------------------------

#define NN    20000
#define EE    240000
#define HH    256
#define NHEAD 8
#define HDIM  32
#define NRBF  64

typedef __attribute__((ext_vector_type(16))) __bf16 v16bf;
typedef __attribute__((ext_vector_type(8)))  __bf16 v8bf;
typedef __attribute__((ext_vector_type(8)))  float  v8f;
typedef __attribute__((ext_vector_type(4)))  unsigned int v4u;
typedef __attribute__((ext_vector_type(8)))  int v8i;
typedef __attribute__((ext_vector_type(4)))  int v4i;

#define ASTR 264   // 256 bf16 + 16B TDM pad -> 528B row stride (16B aligned)

// ---- WMMA fragment helpers (CDNA5 16x16x32 bf16 layouts, wave32) ----------
// A (16x32, row-striped per lane): lane L holds row M=L&15; elements 0..7 are
// K = kbase + (L>=16?8:0) + j, elements 8..15 are K = kbase + 16 + (L>=16?8:0) + j.
static __device__ __forceinline__ v16bf load_a_tile(const __bf16* base, int lda, int lane) {
  int m = lane & 15, hi = lane >> 4;
  const __bf16* p = base + m * lda + hi * 8;
  v8bf c0 = *(const v8bf*)(p);
  v8bf c1 = *(const v8bf*)(p + 16);
  v16bf a;
#pragma unroll
  for (int j = 0; j < 8; ++j) { a[j] = c0[j]; a[j + 8] = c1[j]; }
  return a;
}

// B (32x16) from an N-major (transposed) weight image Wt[n][k]:
// lane L holds column N=L&15; elements 0..7 are K = kbase + (L>=16?8:0) + j,
// elements 8..15 are K = kbase + 16 + (L>=16?8:0) + j.
static __device__ __forceinline__ v16bf load_b_tile(const __bf16* baseT, int ldk, int lane) {
  int n = lane & 15, hi = lane >> 4;
  const __bf16* p = baseT + n * ldk + hi * 8;
  v8bf c0 = *(const v8bf*)(p);
  v8bf c1 = *(const v8bf*)(p + 16);
  v16bf b;
#pragma unroll
  for (int j = 0; j < 8; ++j) { b[j] = c0[j]; b[j + 8] = c1[j]; }
  return b;
}

static __device__ __forceinline__ v8f wmma_bf16(v16bf a, v16bf b, v8f c) {
  return __builtin_amdgcn_wmma_f32_16x16x32_bf16(false, a, false, b, (short)0, c, false, false);
}

// C/D (16x16 f32): element r of lane L is row (r + (L>=16?8:0)), col (L&15).

// ---- Tensor Data Mover: 1-D copy of `ndw` DWORDs global->LDS with pad -----
// Pads 4 DWORDs (16B) after every 128 DWORDs (512B = one 256-bf16 row), so a
// [16][256] bf16 strip lands in LDS with a 528B (ASTR) row stride.
// D# per CDNA5 ISA: group0 = {flags, lds_addr,
// global_addr lo, global_addr hi | type=2}; group1 packs data_size/pad/dims.
static __device__ __forceinline__ void tdm_load_strip(unsigned lds_off, const void* gptr, int ndw) {
  unsigned long long ga = (unsigned long long)(uintptr_t)gptr;
  v4u g0;
  g0[0] = 1u;                                   // count=1, user descriptor
  g0[1] = lds_off;                              // lds_addr (bytes)
  g0[2] = (unsigned)ga;                         // global_addr[31:0]
  g0[3] = (unsigned)((ga >> 32) & 0x01FFFFFFu) | (2u << 30);  // addr[56:32] | type=2
  v8i g1;
  g1[0] = (2 << 16)      // data_size = 4B
        | (1 << 20)      // pad_enable
        | (6 << 22)      // pad_interval: 128 DWORDs
        | (3 << 25);     // pad_amount: 4 DWORDs
  g1[1] = (ndw & 0xFFFF) << 16;                               // tensor_dim0[15:0]
  g1[2] = ((ndw >> 16) & 0xFFFF) | (1 << 16);                 // tensor_dim0[31:16] | tensor_dim1=1
  g1[3] = (ndw & 0xFFFF) << 16;                               // tile_dim0 = ndw
  g1[4] = 0;                                                  // tile_dim1=0, tile_dim2=0
  g1[5] = ndw;                                                // tensor_dim0_stride lo
  g1[6] = 0;                                                  // stride hi / dim1_stride lo
  g1[7] = 0;
  v4i z4 = {};
#if defined(__clang_major__) && (__clang_major__ >= 23)
  v8i z8 = {};
  __builtin_amdgcn_tensor_load_to_lds(g0, g1, z4, z4, z8, 0);
#else
  __builtin_amdgcn_tensor_load_to_lds(g0, g1, z4, z4, 0);
#endif
}

// ---------------------------------------------------------------------------
// Weight convert + transpose: Wt[n*K+k] = bf16(W[k*N+n])
__global__ void wcvt_kernel(const float* __restrict__ w, __bf16* __restrict__ wt, int K, int N) {
  int idx = blockIdx.x * 256 + threadIdx.x;
  if (idx >= K * N) return;
  int n = idx / K, k = idx - n * K;
  wt[idx] = (__bf16)w[k * N + n];
}

__global__ void cvt_f32_bf16_kernel(const float* __restrict__ s, __bf16* __restrict__ d, int total) {
  int idx = blockIdx.x * 256 + threadIdx.x;
  if (idx < total) d[idx] = (__bf16)s[idx];
}

// ---------------------------------------------------------------------------
// LayerNorm: one wave per row (wave32, 8 floats/lane), writes f32 + bf16 copies
__global__ __launch_bounds__(256) void ln_kernel(const float* __restrict__ x,
                                                 const float* __restrict__ ln_w,
                                                 const float* __restrict__ ln_b,
                                                 float* __restrict__ xn,
                                                 __bf16* __restrict__ xnb) {
  int wave = threadIdx.x >> 5, lane = threadIdx.x & 31;
  int row = blockIdx.x * 8 + wave;
  const float* rp = x + row * HH + lane * 8;
  float v[8];
  float4 a0 = *(const float4*)(rp);
  float4 a1 = *(const float4*)(rp + 4);
  v[0] = a0.x; v[1] = a0.y; v[2] = a0.z; v[3] = a0.w;
  v[4] = a1.x; v[5] = a1.y; v[6] = a1.z; v[7] = a1.w;
  float s = 0.f, sq = 0.f;
#pragma unroll
  for (int j = 0; j < 8; ++j) { s += v[j]; sq += v[j] * v[j]; }
#pragma unroll
  for (int off = 16; off > 0; off >>= 1) {
    s  += __shfl_xor(s,  off, 32);
    sq += __shfl_xor(sq, off, 32);
  }
  float mu = s * (1.f / HH);
  float var = sq * (1.f / HH) - mu * mu;
  float rstd = rsqrtf(var + 1e-5f);
#pragma unroll
  for (int j = 0; j < 8; ++j) {
    int c = lane * 8 + j;
    float o = (v[j] - mu) * rstd * ln_w[c] + ln_b[c];
    xn[row * HH + c]  = o;
    xnb[row * HH + c] = (__bf16)o;
  }
}

// ---------------------------------------------------------------------------
// Q/K/V projections: C[N,256] = xnb @ W + b, bf16 WMMA, bf16 output.
// grid = (1250, 3); block = 128 (4 waves); wave computes 4 N-tiles of 16.
// A-strip (16x256 bf16, contiguous 8KB) staged into LDS by TDM with pad.
__global__ __launch_bounds__(128) void qkv_kernel(const __bf16* __restrict__ xnb,
                                                  const __bf16* __restrict__ wqt,
                                                  const __bf16* __restrict__ wkt,
                                                  const __bf16* __restrict__ wvt,
                                                  const float* __restrict__ bq,
                                                  const float* __restrict__ bk,
                                                  const float* __restrict__ bv,
                                                  __bf16* __restrict__ Qb,
                                                  __bf16* __restrict__ Kb,
                                                  __bf16* __restrict__ Vb) {
  __shared__ __align__(16) __bf16 a_lds[16][ASTR];
  int wave = threadIdx.x >> 5, lane = threadIdx.x & 31;
  int m0 = blockIdx.x * 16;
  const __bf16* wt; const float* bias; __bf16* out;
  if (blockIdx.y == 0)      { wt = wqt; bias = bq; out = Qb; }
  else if (blockIdx.y == 1) { wt = wkt; bias = bk; out = Kb; }
  else                      { wt = wvt; bias = bv; out = Vb; }

  if (wave == 0) {   // one TDM issue per workgroup; EXEC-independent, scalar D#
    tdm_load_strip((unsigned)(uintptr_t)&a_lds[0][0], xnb + m0 * HH, (16 * HH * 2) / 4);
    __builtin_amdgcn_s_wait_tensorcnt(0);
  }
  __syncthreads();

  const __bf16* A = &a_lds[0][0];
  for (int t = 0; t < 4; ++t) {
    int n0 = (wave * 4 + t) * 16;
    v8f acc = {};
    for (int kk = 0; kk < 8; ++kk) {
      v16bf a = load_a_tile(A + kk * 32, ASTR, lane);   // ds_read_b128 pairs
      v16bf b = load_b_tile(wt + n0 * HH + kk * 32, HH, lane);
      acc = wmma_bf16(a, b, acc);
    }
    int nl = lane & 15, mb = (lane >> 4) * 8;
    int n = n0 + nl;
    float bn = bias[n];
#pragma unroll
    for (int r = 0; r < 8; ++r)
      out[(m0 + mb + r) * HH + n] = (__bf16)(acc[r] + bn);
  }
}

// ---------------------------------------------------------------------------
// Edge kernel: per wave, 16 edges.
//   phase A: head-softmax attention weights (wave32 shfl over lane pairs)
//   phase B: h = silu(rbf@W1+b1) via WMMA -> LDS (bf16)
//   phase C: w_phys = (h@W2+b2)*cutoff via WMMA (A from LDS), fuse
//            msg_phys + gate*msg_attn, scatter-add to aggr with f32 atomics.
#define HSTR 264  // 256 + 8 bf16 pad (16B) -> bank-decorrelated rows
__global__ __launch_bounds__(128) void edge_kernel(const int* __restrict__ ei,
                                                   const float* __restrict__ rbf,
                                                   const float* __restrict__ cutoff,
                                                   const __bf16* __restrict__ w1t,
                                                   const float* __restrict__ b1,
                                                   const __bf16* __restrict__ w2t,
                                                   const float* __restrict__ b2,
                                                   const float* __restrict__ xn,
                                                   const __bf16* __restrict__ Qb,
                                                   const __bf16* __restrict__ Kb,
                                                   const __bf16* __restrict__ Vb,
                                                   const float* __restrict__ gate,
                                                   float* __restrict__ aggr) {
  __shared__ __align__(16) __bf16 h_lds[4][16][HSTR];
  __shared__ float attn_lds[4][16][NHEAD];
  __shared__ float cut_s[4][16];
  __shared__ int   row_s[4][16], col_s[4][16];

  int wave = threadIdx.x >> 5, lane = threadIdx.x & 31;
  int e0 = (blockIdx.x * 4 + wave) * 16;
  if (lane < 16) {
    int e = e0 + lane;
    row_s[wave][lane] = ei[e];
    col_s[wave][lane] = ei[EE + e];
    cut_s[wave][lane] = cutoff[e];
  }
  __syncthreads();

  float g0 = gate[0];
  int el = lane & 15;      // local edge
  int hg = lane >> 4;      // head group: heads hg*4 .. hg*4+3
  int rown = row_s[wave][el];
  int coln = col_s[wave][el];
  float cut = cut_s[wave][el];

  // warm L2->L0 for the gather rows used in phase C
  __builtin_prefetch(xn + (size_t)coln * HH, 0, 1);
  __builtin_prefetch(Vb + (size_t)coln * HH, 0, 1);

  // ---- phase A: attention (softmax over the 8 heads of each edge) ----
  float sc[4];
#pragma unroll
  for (int hh = 0; hh < 4; ++hh) {
    int head = hg * 4 + hh;
    const __bf16* qp = Qb + rown * HH + head * HDIM;
    const __bf16* kp = Kb + coln * HH + head * HDIM;
    float acc = 0.f;
#pragma unroll
    for (int c = 0; c < 4; ++c) {
      v8bf qv = *(const v8bf*)(qp + c * 8);
      v8bf kv = *(const v8bf*)(kp + c * 8);
#pragma unroll
      for (int j = 0; j < 8; ++j) acc += (float)qv[j] * (float)kv[j];
    }
    sc[hh] = acc * 0.17677669529663687f * cut;  // 1/sqrt(32) * cutoff
  }
  float mx = fmaxf(fmaxf(sc[0], sc[1]), fmaxf(sc[2], sc[3]));
  mx = fmaxf(mx, __shfl_xor(mx, 16, 32));      // pair holds the other 4 heads
  float se = 0.f;
#pragma unroll
  for (int hh = 0; hh < 4; ++hh) { sc[hh] = __expf(sc[hh] - mx); se += sc[hh]; }
  se += __shfl_xor(se, 16, 32);
  float inv = 1.f / se;
#pragma unroll
  for (int hh = 0; hh < 4; ++hh) attn_lds[wave][el][hg * 4 + hh] = sc[hh] * inv;

  // ---- phase B: layer 1 of the RBF MLP (K=64 -> two WMMA k-steps/tile) ----
  int mA = lane & 15, hiA = lane >> 4;
  int nl = lane & 15, mb = (lane >> 4) * 8;
  for (int t = 0; t < 16; ++t) {
    v8f acc = {};
#pragma unroll
    for (int ks = 0; ks < 2; ++ks) {
      // A fragment straight from f32 rbf, converted to bf16 in-register
      const float* p = rbf + (e0 + mA) * NRBF + ks * 32 + hiA * 8;
      v16bf a;
#pragma unroll
      for (int j = 0; j < 8; ++j) { a[j] = (__bf16)p[j]; a[j + 8] = (__bf16)p[16 + j]; }
      v16bf b = load_b_tile(w1t + (t * 16) * NRBF + ks * 32, NRBF, lane);
      acc = wmma_bf16(a, b, acc);
    }
    int n = t * 16 + nl;
    float bn = b1[n];
#pragma unroll
    for (int r = 0; r < 8; ++r) {
      float v = acc[r] + bn;
      v = v / (1.f + __expf(-v));                // SiLU
      h_lds[wave][mb + r][n] = (__bf16)v;
    }
  }
  __syncthreads();

  // ---- phase C: layer 2 + cutoff gate + attn message + scatter-add ----
  for (int t = 0; t < 16; ++t) {
    v8f acc = {};
    for (int kk = 0; kk < 8; ++kk) {
      const __bf16* hp = &h_lds[wave][mA][kk * 32 + hiA * 8];
      v8bf c0 = *(const v8bf*)(hp);
      v8bf c1 = *(const v8bf*)(hp + 16);
      v16bf a;
#pragma unroll
      for (int j = 0; j < 8; ++j) { a[j] = c0[j]; a[j + 8] = c1[j]; }
      v16bf b = load_b_tile(w2t + (t * 16) * HH + kk * 32, HH, lane);
      acc = wmma_bf16(a, b, acc);
    }
    int n = t * 16 + nl;
    int head = n >> 5;
    float b2n = b2[n];
#pragma unroll
    for (int r = 0; r < 8; ++r) {
      int m = mb + r;
      float wp  = (acc[r] + b2n) * cut_s[wave][m];
      int cc    = col_s[wave][m];
      float msg = xn[cc * HH + n] * wp;
      float at  = attn_lds[wave][m][head];
      float mv  = at * (float)Vb[cc * HH + n];
      atomicAdd(&aggr[row_s[wave][m] * HH + n], msg + g0 * mv);
    }
  }
}

// ---------------------------------------------------------------------------
// Output projection: out = x + aggr @ Wo + bo (A-strip TDM-staged like QKV)
__global__ __launch_bounds__(128) void out_kernel(const __bf16* __restrict__ aggrb,
                                                  const __bf16* __restrict__ wot,
                                                  const float* __restrict__ bo,
                                                  const float* __restrict__ x,
                                                  float* __restrict__ out) {
  __shared__ __align__(16) __bf16 a_lds[16][ASTR];
  int wave = threadIdx.x >> 5, lane = threadIdx.x & 31;
  int m0 = blockIdx.x * 16;

  if (wave == 0) {
    tdm_load_strip((unsigned)(uintptr_t)&a_lds[0][0], aggrb + m0 * HH, (16 * HH * 2) / 4);
    __builtin_amdgcn_s_wait_tensorcnt(0);
  }
  __syncthreads();

  const __bf16* A = &a_lds[0][0];
  for (int t = 0; t < 4; ++t) {
    int n0 = (wave * 4 + t) * 16;
    v8f acc = {};
    for (int kk = 0; kk < 8; ++kk) {
      v16bf a = load_a_tile(A + kk * 32, ASTR, lane);
      v16bf b = load_b_tile(wot + n0 * HH + kk * 32, HH, lane);
      acc = wmma_bf16(a, b, acc);
    }
    int nl = lane & 15, mb = (lane >> 4) * 8;
    int n = n0 + nl;
    float bn = bo[n];
#pragma unroll
    for (int r = 0; r < 8; ++r) {
      int m = m0 + mb + r;
      out[m * HH + n] = x[m * HH + n] + bn + acc[r];
    }
  }
}

// ---------------------------------------------------------------------------
extern "C" void kernel_launch(void* const* d_in, const int* in_sizes, int n_in,
                              void* d_out, int out_size, void* d_ws, size_t ws_size,
                              hipStream_t stream) {
  const float* x      = (const float*)d_in[0];
  const int*   ei     = (const int*)d_in[1];
  const float* rbf    = (const float*)d_in[2];
  const float* cutoff = (const float*)d_in[3];
  const float* w1     = (const float*)d_in[4];
  const float* b1     = (const float*)d_in[5];
  const float* w2     = (const float*)d_in[6];
  const float* b2     = (const float*)d_in[7];
  const float* ln_w   = (const float*)d_in[8];
  const float* ln_b   = (const float*)d_in[9];
  const float* wq     = (const float*)d_in[10];
  const float* bq     = (const float*)d_in[11];
  const float* wk     = (const float*)d_in[12];
  const float* bk     = (const float*)d_in[13];
  const float* wv     = (const float*)d_in[14];
  const float* bv     = (const float*)d_in[15];
  const float* wo     = (const float*)d_in[16];
  const float* bo     = (const float*)d_in[17];
  const float* gate   = (const float*)d_in[18];
  float* out = (float*)d_out;

  // workspace carve-up (256B aligned)
  char* ws = (char*)d_ws;
  size_t off = 0;
  auto take = [&](size_t bytes) { char* p = ws + off; off = (off + bytes + 255) & ~(size_t)255; return p; };
  float*  xn    = (float*)take((size_t)NN * HH * 4);
  __bf16* xnb   = (__bf16*)take((size_t)NN * HH * 2);
  __bf16* Qb    = (__bf16*)take((size_t)NN * HH * 2);
  __bf16* Kb    = (__bf16*)take((size_t)NN * HH * 2);
  __bf16* Vb    = (__bf16*)take((size_t)NN * HH * 2);
  float*  aggr  = (float*)take((size_t)NN * HH * 4);
  __bf16* aggrb = (__bf16*)take((size_t)NN * HH * 2);
  __bf16* w1t   = (__bf16*)take((size_t)NRBF * HH * 2);
  __bf16* w2t   = (__bf16*)take((size_t)HH * HH * 2);
  __bf16* wqt   = (__bf16*)take((size_t)HH * HH * 2);
  __bf16* wkt   = (__bf16*)take((size_t)HH * HH * 2);
  __bf16* wvt   = (__bf16*)take((size_t)HH * HH * 2);
  __bf16* wot   = (__bf16*)take((size_t)HH * HH * 2);
  (void)ws_size; (void)n_in; (void)in_sizes; (void)out_size;

  // 1) weight convert+transpose to bf16 N-major images
  int gw = (NRBF * HH + 255) / 256, gh = (HH * HH + 255) / 256;
  wcvt_kernel<<<gw, 256, 0, stream>>>(w1, w1t, NRBF, HH);
  wcvt_kernel<<<gh, 256, 0, stream>>>(w2, w2t, HH, HH);
  wcvt_kernel<<<gh, 256, 0, stream>>>(wq, wqt, HH, HH);
  wcvt_kernel<<<gh, 256, 0, stream>>>(wk, wkt, HH, HH);
  wcvt_kernel<<<gh, 256, 0, stream>>>(wv, wvt, HH, HH);
  wcvt_kernel<<<gh, 256, 0, stream>>>(wo, wot, HH, HH);

  // 2) LayerNorm (f32 + bf16 copies)
  ln_kernel<<<NN / 8, 256, 0, stream>>>(x, ln_w, ln_b, xn, xnb);

  // 3) zero the scatter accumulator (graph-capturable memset node)
  hipMemsetAsync(aggr, 0, (size_t)NN * HH * 4, stream);

  // 4) Q/K/V projections (WMMA, TDM-staged A)
  qkv_kernel<<<dim3(NN / 16, 3), 128, 0, stream>>>(xnb, wqt, wkt, wvt, bq, bk, bv, Qb, Kb, Vb);

  // 5) fused edge pipeline (WMMA MLP + attention + scatter atomics)
  edge_kernel<<<EE / 64, 128, 0, stream>>>(ei, rbf, cutoff, w1t, b1, w2t, b2,
                                           xn, Qb, Kb, Vb, gate, aggr);

  // 6) aggr -> bf16 for the final WMMA projection
  cvt_f32_bf16_kernel<<<(NN * HH + 255) / 256, 256, 0, stream>>>(aggr, aggrb, NN * HH);

  // 7) out = x + aggr @ Wo + bo (WMMA, TDM-staged A)
  out_kernel<<<NN / 16, 128, 0, stream>>>(aggrb, wot, bo, x, out);
}
